// BEVFormerHead_79551384256541
// MI455X (gfx1250) — matrix operand order repaired and checked by reading, not compiled, not run on hardware
//
#include <hip/hip_runtime.h>
#include <hip/hip_bf16.h>

#define EDIM 256
#define HEADS 8
#define HD 32
#define NPT 4
#define NCAMS 6
#define BEVW 128
#define NBEV (BEVW*BEVW)
#define NQD 900
#define NQP 928          // 900 padded to multiple of 32
#define FFND 512
#define NFEAT 30825
#define NFEATP 30848     // padded to multiple of 32

typedef __attribute__((ext_vector_type(16))) _Float16 v16h;
typedef __attribute__((ext_vector_type(8)))  float    v8f;

// ---------------------------------------------------------------------------
// Weight repack: fp32 W[K][N] -> f16 in WMMA B-fragment order.
// For tile (kt, nt):  Wp[(kt*(N/16)+nt)*512 + lane*16 + j]
//                       = W[kt*32 + (lane>=16)*8 + (j&7) + (j>>3)*16][nt*16 + (lane&15)]
// so in the GEMM each lane reads its 16 halves as one aligned 32B vector.
// ---------------------------------------------------------------------------
__global__ void repack_w_k(const float* __restrict__ W, _Float16* __restrict__ Wp,
                           int K, int N) {
  int idx = blockIdx.x * blockDim.x + threadIdx.x;
  if (idx >= K * N) return;
  int tile = idx >> 9, within = idx & 511;
  int lane = within >> 4, j = within & 15;
  int ntn = N >> 4;
  int kt = tile / ntn, nt = tile - kt * ntn;
  int krow = kt * 32 + (lane >> 4) * 8 + (j & 7) + (j >> 3) * 16;
  int ncol = nt * 16 + (lane & 15);
  Wp[idx] = (_Float16)W[(size_t)krow * N + ncol];
}

// A 16x32 f16 fragment (ISA 7.12.2): lane&15 = M row; lanes 0-15 hold
// K {0..7,16..23}, lanes 16-31 hold K {8..15,24..31}.
__device__ __forceinline__ v16h load_a_frag(const float* __restrict__ xk) {
  float4 A0 = *(const float4*)(xk);
  float4 A1 = *(const float4*)(xk + 4);
  float4 A2 = *(const float4*)(xk + 16);
  float4 A3 = *(const float4*)(xk + 20);
  v16h a;
  a[0]  = (_Float16)A0.x; a[1]  = (_Float16)A0.y;
  a[2]  = (_Float16)A0.z; a[3]  = (_Float16)A0.w;
  a[4]  = (_Float16)A1.x; a[5]  = (_Float16)A1.y;
  a[6]  = (_Float16)A1.z; a[7]  = (_Float16)A1.w;
  a[8]  = (_Float16)A2.x; a[9]  = (_Float16)A2.y;
  a[10] = (_Float16)A2.z; a[11] = (_Float16)A2.w;
  a[12] = (_Float16)A3.x; a[13] = (_Float16)A3.y;
  a[14] = (_Float16)A3.z; a[15] = (_Float16)A3.w;
  return a;
}

// ---------------------------------------------------------------------------
// WMMA GEMM:  Y[M,N] = act(X[M,K] @ W[K,N] + bias)
// One wave computes a 32 x (16*NT) tile: per k-step, 2 A fragments and NT B
// fragments are loaded up front (one clause), then 2*NT independent
// v_wmma_f32_16x16x32_f16 run back-to-back.  M multiple of 32 (callers pad;
// pad rows only produce pad rows).  K multiple of 32, N of 16*NT.
// ---------------------------------------------------------------------------
template<int ACT, int NT>
__global__ void gemm_wmma(const float* __restrict__ X, const _Float16* __restrict__ Wp,
                          const float* __restrict__ bias, float* __restrict__ Y,
                          int M, int N, int K) {
  const int tm  = blockIdx.x * 32;
  const int tn0 = blockIdx.y * 16 * NT;
  const int lane = threadIdx.x;
  const int half = lane >> 4, l16 = lane & 15;
  const float* xr0 = X + (size_t)(tm + l16) * K + half * 8;
  const float* xr1 = xr0 + (size_t)16 * K;
  const int ntn = N >> 4;
  v8f c0[NT], c1[NT];
#pragma unroll
  for (int nt = 0; nt < NT; ++nt) { c0[nt] = (v8f){}; c1[nt] = (v8f){}; }
#pragma unroll 2
  for (int k0 = 0; k0 < K; k0 += 32) {
    v16h a0 = load_a_frag(xr0 + k0);
    v16h a1 = load_a_frag(xr1 + k0);
    const _Float16* wp = Wp + ((size_t)(k0 >> 5) * ntn + (tn0 >> 4)) * 512 + lane * 16;
    v16h bf[NT];
#pragma unroll
    for (int nt = 0; nt < NT; ++nt) bf[nt] = *(const v16h*)(wp + (size_t)nt * 512);
#pragma unroll
    for (int nt = 0; nt < NT; ++nt)
      c0[nt] = __builtin_amdgcn_wmma_f32_16x16x32_f16(false, a0, false, bf[nt],
                                                      (short)0, c0[nt], false, false);
#pragma unroll
    for (int nt = 0; nt < NT; ++nt)
      c1[nt] = __builtin_amdgcn_wmma_f32_16x16x32_f16(false, a1, false, bf[nt],
                                                      (short)0, c1[nt], false, false);
  }
  const int row0 = tm + half * 8;
#pragma unroll
  for (int nt = 0; nt < NT; ++nt) {
    const int col = tn0 + nt * 16 + l16;
    const float bv = bias[col];
#pragma unroll
    for (int r = 0; r < 8; ++r) {
      float v = c0[nt][r] + bv;
      if (ACT == 1) v = fmaxf(v, 0.f);
      Y[(size_t)(row0 + r) * N + col] = v;
    }
#pragma unroll
    for (int r = 0; r < 8; ++r) {
      float v = c1[nt][r] + bv;
      if (ACT == 1) v = fmaxf(v, 0.f);
      Y[(size_t)(row0 + 16 + r) * N + col] = v;
    }
  }
}

__global__ void add_k(const float* __restrict__ a, const float* __restrict__ b,
                      float* __restrict__ y, int n) {
  int i = blockIdx.x * blockDim.x + threadIdx.x;
  if (i < n) y[i] = a[i] + b[i];
}

// LayerNorm(x + r) over rows of 256.  One wave per row.
__global__ void ln_res_k(const float* __restrict__ x, const float* __restrict__ r,
                         const float* __restrict__ g, const float* __restrict__ b,
                         float* __restrict__ y) {
  const int row = blockIdx.x, lane = threadIdx.x;
  const float* xr = x + (size_t)row * EDIM;
  const float* rr = r + (size_t)row * EDIM;
  float v[8];
  float s = 0.f;
#pragma unroll
  for (int i = 0; i < 8; ++i) {
    float t = xr[i * 32 + lane] + rr[i * 32 + lane];
    v[i] = t; s += t;
  }
  for (int o = 16; o; o >>= 1) s += __shfl_xor(s, o);
  const float mean = s * (1.f / EDIM);
  float vs = 0.f;
#pragma unroll
  for (int i = 0; i < 8; ++i) { float d = v[i] - mean; vs += d * d; }
  for (int o = 16; o; o >>= 1) vs += __shfl_xor(vs, o);
  const float rstd = rsqrtf(vs * (1.f / EDIM) + 1e-5f);
#pragma unroll
  for (int i = 0; i < 8; ++i) {
    int e = i * 32 + lane;
    y[(size_t)row * EDIM + e] = (v[i] - mean) * rstd * g[e] + b[e];
  }
}

// Softmax over last dim S (<=16), in place.
__global__ void softmax_k(float* __restrict__ x, int rows, int S) {
  int r = blockIdx.x * blockDim.x + threadIdx.x;
  if (r >= rows) return;
  float* p = x + (size_t)r * S;
  float m = -1e30f;
  for (int i = 0; i < S; ++i) m = fmaxf(m, p[i]);
  float e[16], z = 0.f;
  for (int i = 0; i < S; ++i) { e[i] = __expf(p[i] - m); z += e[i]; }
  float inv = 1.f / z;
  for (int i = 0; i < S; ++i) p[i] = e[i] * inv;
}

// ---------------------------------------------------------------------------
// Multi-scale deformable sampling.  One thread per (q, head, 4-channel quad):
// all gathers are b128.  out = (accum ? out : 0) + scale * sum_l,p aw * bilerp
// ---------------------------------------------------------------------------
struct DefShapes { int h[4]; int w[4]; int voff[4]; float nw[4]; float nh[4]; int L; };

__global__ void ms_deform_k(const float* __restrict__ value, const float* __restrict__ off,
                            const float* __restrict__ aw, const float* __restrict__ refpts,
                            int ref_stride, float* __restrict__ out, int Nq,
                            DefShapes sh, float scale, int accum) {
  int idx = blockIdx.x * blockDim.x + threadIdx.x;
  if (idx >= Nq * 64) return;
  const int quad = (idx & 7) * 4;
  const int h = (idx >> 3) & 7;
  const int q = idx >> 6;
  float rx, ry;
  if (refpts) { rx = refpts[(size_t)q * ref_stride]; ry = refpts[(size_t)q * ref_stride + 1]; }
  else { rx = ((q & (BEVW - 1)) + 0.5f) * (1.f / BEVW); ry = ((q >> 7) + 0.5f) * (1.f / BEVW); }
  const int LP = sh.L * NPT;
  const float* offq = off + (size_t)(q * HEADS + h) * LP * 2;
  const float* awq  = aw  + (size_t)(q * HEADS + h) * LP;
  float ax = 0.f, ay = 0.f, az = 0.f, aww = 0.f;
  for (int l = 0; l < sh.L; ++l) {
    const int W = sh.w[l], Hh = sh.h[l];
    const float* vbase = value + ((size_t)sh.voff[l]) * EDIM + h * HD + quad;
    const float invw = 1.f / sh.nw[l], invh = 1.f / sh.nh[l];
    for (int p = 0; p < NPT; ++p) {
      const int ip = l * NPT + p;
      const float lx = rx + offq[ip * 2 + 0] * invw;
      const float ly = ry + offq[ip * 2 + 1] * invh;
      const float xf = lx * W - 0.5f, yf = ly * Hh - 0.5f;
      const float x0f = floorf(xf), y0f = floorf(yf);
      const int x0 = (int)x0f, y0 = (int)y0f;
      const float wx = xf - x0f, wy = yf - y0f;
      float sx = 0.f, sy = 0.f, sz = 0.f, sw = 0.f;
      auto tap = [&](int xi, int yi, float wgt) {
        if (xi >= 0 && xi < W && yi >= 0 && yi < Hh) {
          float4 v = *(const float4*)(vbase + (size_t)(yi * W + xi) * EDIM);
          sx += wgt * v.x; sy += wgt * v.y; sz += wgt * v.z; sw += wgt * v.w;
        }
      };
      tap(x0,     y0,     (1.f - wx) * (1.f - wy));
      tap(x0 + 1, y0,     wx * (1.f - wy));
      tap(x0,     y0 + 1, (1.f - wx) * wy);
      tap(x0 + 1, y0 + 1, wx * wy);
      const float a = awq[ip];
      ax += a * sx; ay += a * sy; az += a * sz; aww += a * sw;
    }
  }
  float* op = out + (size_t)q * EDIM + h * HD + quad;
  if (accum) {
    op[0] += scale * ax; op[1] += scale * ay; op[2] += scale * az; op[3] += scale * aww;
  } else {
    op[0] = scale * ax; op[1] = scale * ay; op[2] = scale * az; op[3] = scale * aww;
  }
}

// Decoder MHA.  One wave per (q, head); lane = d.
__global__ void mha_k(const float* __restrict__ Q, const float* __restrict__ K,
                      const float* __restrict__ V, float* __restrict__ O, int Nq) {
  __shared__ float sc[4][904];
  const int wid = threadIdx.x >> 5, lane = threadIdx.x & 31;
  const int task = blockIdx.x * 4 + wid;
  const int h = task & 7, q = task >> 3;
  const float qv = Q[(size_t)q * EDIM + h * HD + lane];
  float mx = -1e30f;
  for (int k = 0; k < Nq; ++k) {
    float s = qv * K[(size_t)k * EDIM + h * HD + lane];
    for (int o = 16; o; o >>= 1) s += __shfl_xor(s, o);
    s *= 0.17677669529663687f;          // 1/sqrt(32)
    if (lane == 0) sc[wid][k] = s;
    mx = fmaxf(mx, s);
  }
  __syncthreads();
  float z = 0.f;
  for (int k = lane; k < Nq; k += 32) {
    float e = __expf(sc[wid][k] - mx);
    sc[wid][k] = e; z += e;
  }
  for (int o = 16; o; o >>= 1) z += __shfl_xor(z, o);
  __syncthreads();
  const float inv = 1.f / z;
  float acc = 0.f;
  for (int k = 0; k < Nq; ++k) acc += sc[wid][k] * V[(size_t)k * EDIM + h * HD + lane];
  O[(size_t)q * EDIM + h * HD + lane] = acc * inv;
}

// can_bus MLP (18->128 relu ->256 relu) + LayerNorm.  One 256-thread block.
__global__ void canbus_k(const float* __restrict__ can,
                         const float* __restrict__ w1, const float* __restrict__ b1,
                         const float* __restrict__ w2, const float* __restrict__ b2,
                         const float* __restrict__ lng, const float* __restrict__ lnb,
                         float* __restrict__ cb) {
  __shared__ float h1[128];
  __shared__ float red[8];
  const int t = threadIdx.x;
  if (t < 128) {
    float s = b1[t];
    for (int i = 0; i < 18; ++i) s += can[i] * w1[i * 128 + t];
    h1[t] = fmaxf(s, 0.f);
  }
  __syncthreads();
  float s = b2[t];
  for (int i = 0; i < 128; ++i) s += h1[i] * w2[i * EDIM + t];
  s = fmaxf(s, 0.f);
  float ps = s;
  for (int o = 16; o; o >>= 1) ps += __shfl_xor(ps, o);
  if ((t & 31) == 0) red[t >> 5] = ps;
  __syncthreads();
  float tot = 0.f;
  for (int i = 0; i < 8; ++i) tot += red[i];
  const float mean = tot * (1.f / EDIM);
  __syncthreads();
  float d = s - mean, pv = d * d;
  for (int o = 16; o; o >>= 1) pv += __shfl_xor(pv, o);
  if ((t & 31) == 0) red[t >> 5] = pv;
  __syncthreads();
  float vtot = 0.f;
  for (int i = 0; i < 8; ++i) vtot += red[i];
  cb[t] = d * rsqrtf(vtot * (1.f / EDIM) + 1e-5f) * lng[t] + lnb[t];
}

// bev = bev_queries + cb ;  posf[q][e] = bev_pos[e*NBEV + q]
__global__ void bev_init_k(const float* __restrict__ bq, const float* __restrict__ cb,
                           const float* __restrict__ bev_pos,
                           float* __restrict__ bev, float* __restrict__ posf) {
  int i = blockIdx.x * blockDim.x + threadIdx.x;
  if (i >= NBEV * EDIM) return;
  int e = i & 255, q = i >> 8;
  bev[i]  = bq[i] + cb[e];
  posf[i] = bev_pos[(size_t)e * NBEV + q];
}

// featc[s][e] = feat[cam][e][s] + cam_emb[e] + lvl_emb[e]   (single level)
__global__ void featc_k(const float* __restrict__ feat, const float* __restrict__ cam_emb,
                        const float* __restrict__ lvl_emb, float* __restrict__ dst,
                        int HW, int cam) {
  int i = blockIdx.x * blockDim.x + threadIdx.x;
  if (i >= HW * EDIM) return;
  int e = i & 255, s = i >> 8;
  dst[i] = feat[((size_t)cam * EDIM + e) * HW + s] + cam_emb[e] + lvl_emb[e];
}

__global__ void split_oqe_k(const float* __restrict__ oqe,
                            float* __restrict__ qp, float* __restrict__ q) {
  int i = blockIdx.x * blockDim.x + threadIdx.x;
  if (i >= NQD * EDIM) return;
  int r = i >> 8, e = i & 255;
  qp[i] = oqe[(size_t)r * 512 + e];
  q[i]  = oqe[(size_t)r * 512 + EDIM + e];
}

// ref = sigmoid(qp @ Wref + bref),  N = 3
__global__ void ref_k(const float* __restrict__ qp, const float* __restrict__ w,
                      const float* __restrict__ b, float* __restrict__ out) {
  int i = blockIdx.x * blockDim.x + threadIdx.x;
  if (i >= NQD * 3) return;
  int q = i / 3, j = i % 3;
  float s = b[j];
  const float* qr = qp + (size_t)q * EDIM;
  for (int k = 0; k < EDIM; ++k) s += qr[k] * w[k * 3 + j];
  out[i] = 1.f / (1.f + __expf(-s));
}

// ---------------------------------------------------------------------------
// Host side.  Params flattened as JAX pytree (sorted dict keys; b before w/g).
// ---------------------------------------------------------------------------
struct Lin { const float* w; const float* b; };
struct LNp { const float* g; const float* b; };
struct DecL { Lin ca_attn, ca_off, ca_out, ca_v, ffn1, ffn2; LNp ln1, ln2, ln3;
              Lin wk, wo, wq, wv; };
struct EncL { Lin ffn1, ffn2; LNp ln1, ln2, ln3;
              Lin sca_attn, sca_off, sca_out, sca_v, tsa_attn, tsa_off, tsa_out, tsa_v; };
struct PackedEnc { _Float16 *tsa_v,*tsa_off,*tsa_attn,*tsa_out,*sca_off,*sca_attn,
                            *sca_v,*sca_out,*ffn1,*ffn2; };
struct PackedDec { _Float16 *wq,*wk,*wv,*wo,*ca_v,*ca_off,*ca_attn,*ca_out,*ffn1,*ffn2; };

static Lin getLin(void* const* d_in, int& i) {
  Lin l; l.b = (const float*)d_in[i++]; l.w = (const float*)d_in[i++]; return l;
}
static LNp getLN(void* const* d_in, int& i) {
  LNp l; l.b = (const float*)d_in[i++]; l.g = (const float*)d_in[i++]; return l;
}

extern "C" void kernel_launch(void* const* d_in, const int* in_sizes, int n_in,
                              void* d_out, int out_size, void* d_ws, size_t ws_size,
                              hipStream_t stream) {
  (void)in_sizes; (void)n_in; (void)out_size; (void)ws_size;
  const float* feat[4] = { (const float*)d_in[0], (const float*)d_in[1],
                           (const float*)d_in[2], (const float*)d_in[3] };
  const float* bev_queries = (const float*)d_in[4];
  const float* oqe         = (const float*)d_in[5];
  const float* bev_pos     = (const float*)d_in[6];
  const float* can_bus     = (const float*)d_in[7];
  int i = 8;
  const float* cams_embeds = (const float*)d_in[i++];
  Lin cb1 = getLin(d_in, i);
  Lin cb2 = getLin(d_in, i);
  LNp cb_ln = getLN(d_in, i);
  DecL dec[3];
  for (int l = 0; l < 3; ++l) {
    dec[l].ca_attn = getLin(d_in, i); dec[l].ca_off = getLin(d_in, i);
    dec[l].ca_out  = getLin(d_in, i); dec[l].ca_v   = getLin(d_in, i);
    dec[l].ffn1    = getLin(d_in, i); dec[l].ffn2   = getLin(d_in, i);
    dec[l].ln1 = getLN(d_in, i); dec[l].ln2 = getLN(d_in, i); dec[l].ln3 = getLN(d_in, i);
    dec[l].wk = getLin(d_in, i); dec[l].wo = getLin(d_in, i);
    dec[l].wq = getLin(d_in, i); dec[l].wv = getLin(d_in, i);
  }
  EncL enc[2];
  for (int l = 0; l < 2; ++l) {
    enc[l].ffn1 = getLin(d_in, i); enc[l].ffn2 = getLin(d_in, i);
    enc[l].ln1 = getLN(d_in, i); enc[l].ln2 = getLN(d_in, i); enc[l].ln3 = getLN(d_in, i);
    enc[l].sca_attn = getLin(d_in, i); enc[l].sca_off = getLin(d_in, i);
    enc[l].sca_out  = getLin(d_in, i); enc[l].sca_v   = getLin(d_in, i);
    enc[l].tsa_attn = getLin(d_in, i); enc[l].tsa_off = getLin(d_in, i);
    enc[l].tsa_out  = getLin(d_in, i); enc[l].tsa_v   = getLin(d_in, i);
  }
  const float* level_embeds = (const float*)d_in[i++];
  Lin refl = getLin(d_in, i);

  // ---- workspace (floats; every alloc rounded to 8 floats = 32B align) ----
  float* Wsp = (float*)d_ws;
  size_t off = 0;
  auto alloc = [&](size_t n) { n = (n + 7) & ~(size_t)7; float* p = Wsp + off; off += n; return p; };
  float* posf = alloc((size_t)NBEV * EDIM);
  float* qbuf = alloc((size_t)NBEV * EDIM);
  float* vbuf = alloc((size_t)NFEATP * EDIM);   // value buffer (tsa/sca/ca), padded
  float* offb = alloc((size_t)NBEV * 256);
  float* awb  = alloc((size_t)NBEV * 128);
  float* accb = alloc((size_t)NBEV * EDIM);
  float* hbuf = alloc((size_t)NBEV * FFND);     // ffn hidden / per-cam features (padded fits)
  float* obuf = alloc((size_t)NBEV * EDIM);
  float* cb   = alloc(256);
  float* dqp  = alloc((size_t)NQP * EDIM);
  float* dqd  = alloc((size_t)NQP * EDIM);
  float* dx   = alloc((size_t)NQP * EDIM);
  float* dQ   = alloc((size_t)NQP * EDIM);
  float* dK   = alloc((size_t)NQP * EDIM);
  float* dV   = alloc((size_t)NQP * EDIM);
  float* do1  = alloc((size_t)NQP * EDIM);
  float* doff = alloc((size_t)NQP * 64);
  float* dawb = alloc((size_t)NQP * 32);
  float* dref = alloc((size_t)NQD * 3);
  float* dhid = alloc((size_t)NQP * FFND);

  float* bev = (float*)d_out;
  float* out_is  = (float*)d_out + (size_t)NBEV * EDIM;
  float* out_ir0 = out_is + (size_t)3 * NQD * EDIM;
  float* out_irs = out_ir0 + (size_t)NQD * 3;

  // ---- pack all WMMA weights to f16 fragment order (once per launch) ----
  auto packW = [&](const Lin& L, int K, int N) -> _Float16* {
    size_t nh = (size_t)K * N;
    _Float16* p = (_Float16*)alloc((nh + 1) / 2);
    repack_w_k<<<((int)nh + 255) / 256, 256, 0, stream>>>(L.w, p, K, N);
    return p;
  };
  PackedEnc pe[2];
  for (int l = 0; l < 2; ++l) {
    pe[l].tsa_v    = packW(enc[l].tsa_v,    EDIM, EDIM);
    pe[l].tsa_off  = packW(enc[l].tsa_off,  EDIM, 64);
    pe[l].tsa_attn = packW(enc[l].tsa_attn, EDIM, 32);
    pe[l].tsa_out  = packW(enc[l].tsa_out,  EDIM, EDIM);
    pe[l].sca_off  = packW(enc[l].sca_off,  EDIM, 256);
    pe[l].sca_attn = packW(enc[l].sca_attn, EDIM, 128);
    pe[l].sca_v    = packW(enc[l].sca_v,    EDIM, EDIM);
    pe[l].sca_out  = packW(enc[l].sca_out,  EDIM, EDIM);
    pe[l].ffn1     = packW(enc[l].ffn1,     EDIM, FFND);
    pe[l].ffn2     = packW(enc[l].ffn2,     FFND, EDIM);
  }
  PackedDec pd[3];
  for (int l = 0; l < 3; ++l) {
    pd[l].wq      = packW(dec[l].wq,      EDIM, EDIM);
    pd[l].wk      = packW(dec[l].wk,      EDIM, EDIM);
    pd[l].wv      = packW(dec[l].wv,      EDIM, EDIM);
    pd[l].wo      = packW(dec[l].wo,      EDIM, EDIM);
    pd[l].ca_v    = packW(dec[l].ca_v,    EDIM, EDIM);
    pd[l].ca_off  = packW(dec[l].ca_off,  EDIM, 64);
    pd[l].ca_attn = packW(dec[l].ca_attn, EDIM, 32);
    pd[l].ca_out  = packW(dec[l].ca_out,  EDIM, EDIM);
    pd[l].ffn1    = packW(dec[l].ffn1,    EDIM, FFND);
    pd[l].ffn2    = packW(dec[l].ffn2,    FFND, EDIM);
  }

  auto gemm = [&](const float* X, const _Float16* Wp, const float* bias, float* Y,
                  int M, int N, int K, int act) {
    if (N % 64 == 0) {
      dim3 g(M / 32, N / 64);
      if (act) gemm_wmma<1, 4><<<g, 32, 0, stream>>>(X, Wp, bias, Y, M, N, K);
      else     gemm_wmma<0, 4><<<g, 32, 0, stream>>>(X, Wp, bias, Y, M, N, K);
    } else {
      dim3 g(M / 32, N / 32);
      if (act) gemm_wmma<1, 2><<<g, 32, 0, stream>>>(X, Wp, bias, Y, M, N, K);
      else     gemm_wmma<0, 2><<<g, 32, 0, stream>>>(X, Wp, bias, Y, M, N, K);
    }
  };
  auto vadd = [&](const float* a, const float* b, float* y, int n) {
    add_k<<<(n + 255) / 256, 256, 0, stream>>>(a, b, y, n);
  };
  auto lnres = [&](const float* x, const float* r, const LNp& L, float* y, int M) {
    ln_res_k<<<M, 32, 0, stream>>>(x, r, L.g, L.b, y);
  };

  DefShapes sh_bev; sh_bev.L = 1;
  sh_bev.h[0] = BEVW; sh_bev.w[0] = BEVW; sh_bev.voff[0] = 0;
  sh_bev.nw[0] = (float)BEVW; sh_bev.nh[0] = (float)BEVW;
  DefShapes sh_ms; sh_ms.L = 4;
  const int lvl_h[4] = {116, 58, 29, 15}, lvl_w[4] = {200, 100, 50, 25};
  const int lvl_off[4] = {0, 23200, 29000, 30450};
  for (int l = 0; l < 4; ++l) {
    sh_ms.h[l] = lvl_h[l]; sh_ms.w[l] = lvl_w[l]; sh_ms.voff[l] = lvl_off[l];
    sh_ms.nw[l] = (float)lvl_w[l]; sh_ms.nh[l] = (float)lvl_h[l];
  }

  // ---- prologue ----
  canbus_k<<<1, 256, 0, stream>>>(can_bus, cb1.w, cb1.b, cb2.w, cb2.b,
                                  cb_ln.g, cb_ln.b, cb);
  bev_init_k<<<(NBEV * EDIM + 255) / 256, 256, 0, stream>>>(bev_queries, cb, bev_pos,
                                                            bev, posf);

  // ---- encoder ----
  for (int l = 0; l < 2; ++l) {
    const EncL& L = enc[l]; const PackedEnc& P = pe[l];
    // TSA
    vadd(bev, posf, qbuf, NBEV * EDIM);
    gemm(bev,  P.tsa_v,    L.tsa_v.b,    vbuf, NBEV, EDIM, EDIM, 0);
    gemm(qbuf, P.tsa_off,  L.tsa_off.b,  offb, NBEV, 64,   EDIM, 0);
    gemm(qbuf, P.tsa_attn, L.tsa_attn.b, awb,  NBEV, 32,   EDIM, 0);
    softmax_k<<<(NBEV * HEADS + 255) / 256, 256, 0, stream>>>(awb, NBEV * HEADS, NPT);
    ms_deform_k<<<(NBEV * 64 + 255) / 256, 256, 0, stream>>>(vbuf, offb, awb, nullptr, 0,
                                                             accb, NBEV, sh_bev, 1.f, 0);
    gemm(accb, P.tsa_out, L.tsa_out.b, obuf, NBEV, EDIM, EDIM, 0);
    lnres(bev, obuf, L.ln1, bev, NBEV);
    // SCA
    vadd(bev, posf, qbuf, NBEV * EDIM);
    gemm(qbuf, P.sca_off,  L.sca_off.b,  offb, NBEV, 256, EDIM, 0);
    gemm(qbuf, P.sca_attn, L.sca_attn.b, awb,  NBEV, 128, EDIM, 0);
    softmax_k<<<(NBEV * HEADS + 255) / 256, 256, 0, stream>>>(awb, NBEV * HEADS, 16);
    for (int cam = 0; cam < NCAMS; ++cam) {
      for (int lv = 0; lv < 4; ++lv) {
        int HW = lvl_h[lv] * lvl_w[lv], n = HW * EDIM;
        featc_k<<<(n + 255) / 256, 256, 0, stream>>>(feat[lv], cams_embeds + cam * EDIM,
                                                     level_embeds + lv * EDIM,
                                                     hbuf + (size_t)lvl_off[lv] * EDIM,
                                                     HW, cam);
      }
      gemm(hbuf, P.sca_v, L.sca_v.b, vbuf, NFEATP, EDIM, EDIM, 0);
      ms_deform_k<<<(NBEV * 64 + 255) / 256, 256, 0, stream>>>(vbuf, offb, awb, nullptr, 0,
                                                               accb, NBEV, sh_ms,
                                                               1.f / NCAMS, cam > 0);
    }
    gemm(accb, P.sca_out, L.sca_out.b, obuf, NBEV, EDIM, EDIM, 0);
    lnres(bev, obuf, L.ln2, bev, NBEV);
    // FFN
    gemm(bev,  P.ffn1, L.ffn1.b, hbuf, NBEV, FFND, EDIM, 1);
    gemm(hbuf, P.ffn2, L.ffn2.b, obuf, NBEV, EDIM, FFND, 0);
    lnres(bev, obuf, L.ln3, bev, NBEV);
  }

  // ---- decoder prologue ----
  split_oqe_k<<<(NQD * EDIM + 255) / 256, 256, 0, stream>>>(oqe, dqp, dqd);
  ref_k<<<(NQD * 3 + 255) / 256, 256, 0, stream>>>(dqp, refl.w, refl.b, dref);
  hipMemcpyAsync(out_ir0, dref, (size_t)NQD * 3 * sizeof(float),
                 hipMemcpyDeviceToDevice, stream);

  // ---- decoder ----
  for (int l = 0; l < 3; ++l) {
    const DecL& L = dec[l]; const PackedDec& P = pd[l];
    // self attention
    vadd(dqd, dqp, dx, NQD * EDIM);
    gemm(dx,  P.wq, L.wq.b, dQ, NQP, EDIM, EDIM, 0);
    gemm(dx,  P.wk, L.wk.b, dK, NQP, EDIM, EDIM, 0);
    gemm(dqd, P.wv, L.wv.b, dV, NQP, EDIM, EDIM, 0);
    mha_k<<<(NQD * HEADS) / 4, 128, 0, stream>>>(dQ, dK, dV, do1, NQD);
    gemm(do1, P.wo, L.wo.b, dQ, NQP, EDIM, EDIM, 0);
    lnres(dqd, dQ, L.ln1, dqd, NQD);
    // deformable cross attention into BEV
    vadd(dqd, dqp, dx, NQD * EDIM);
    gemm(bev, P.ca_v,    L.ca_v.b,    vbuf, NBEV, EDIM, EDIM, 0);
    gemm(dx,  P.ca_off,  L.ca_off.b,  doff, NQP, 64, EDIM, 0);
    gemm(dx,  P.ca_attn, L.ca_attn.b, dawb, NQP, 32, EDIM, 0);
    softmax_k<<<(NQD * HEADS + 255) / 256, 256, 0, stream>>>(dawb, NQD * HEADS, NPT);
    ms_deform_k<<<(NQD * 64 + 255) / 256, 256, 0, stream>>>(vbuf, doff, dawb, dref, 3,
                                                            do1, NQD, sh_bev, 1.f, 0);
    gemm(do1, P.ca_out, L.ca_out.b, dK, NQP, EDIM, EDIM, 0);
    lnres(dqd, dK, L.ln2, dqd, NQD);
    // FFN
    gemm(dqd,  P.ffn1, L.ffn1.b, dhid, NQP, FFND, EDIM, 1);
    gemm(dhid, P.ffn2, L.ffn2.b, dV,   NQP, EDIM, FFND, 0);
    lnres(dqd, dV, L.ln3, dqd, NQD);
    // outputs
    hipMemcpyAsync(out_is + (size_t)l * NQD * EDIM, dqd,
                   (size_t)NQD * EDIM * sizeof(float), hipMemcpyDeviceToDevice, stream);
    hipMemcpyAsync(out_irs + (size_t)l * NQD * 3, dref,
                   (size_t)NQD * 3 * sizeof(float), hipMemcpyDeviceToDevice, stream);
  }
}
